// FusionTrans_61117384622825
// MI455X (gfx1250) — matrix operand order repaired and checked by reading, not compiled, not run on hardware
//
#include <hip/hip_runtime.h>

typedef __attribute__((ext_vector_type(16))) _Float16 v16h;
typedef __attribute__((ext_vector_type(8)))  float    v8f;

#define MHEADS 6
#define BTOK   65536
#define NEXP   8
#define DF     30
#define DPAD   32
#define EPSV   2.220446049250313e-16f

// Each block: 4 waves, each wave owns one 16-token tile of one head.
// 1024 blocks per head * 4 waves * 16 tokens = 65536 tokens.
// Bias is folded into the GEMM: A column d=30 is constant 1.0 and the
// expert B fragments carry expert_b in row d=30, so WMMA C starts at
// inline 0.  The router is one more WMMA sharing the same A fragment
// (its B rows d>=30 are zero so the bias column is inert).
__global__ __launch_bounds__(128) void moe_heads_kernel(
    const float* __restrict__ xs,        // [M,B,D]
    const float* __restrict__ router_w,  // [M,E,D]
    const float* __restrict__ expert_w,  // [M,E,D(out),D(in)]
    const float* __restrict__ expert_b,  // [M,E,D]
    float* __restrict__ out)             // [M,B,D]
{
  // B-matrix fragments in exact WMMA lane layout: [e][otile][lane][k] f16
  __shared__ __align__(32) _Float16 lds_wB[NEXP * 2 * 32 * 16];   // 16 KB
  __shared__ __align__(32) _Float16 lds_rB[32 * 16];              // 1 KB
  __shared__ __align__(16) float lds_logits[4][16][NEXP];         // 2 KB
  __shared__ float lds_stage[4][16][DPAD];                        // 8 KB

  const int tid  = threadIdx.x;
  const int lane = tid & 31;
  const int wave = tid >> 5;
  const int m    = blockIdx.x >> 10;        // 1024 blocks per head
  const int b0   = ((blockIdx.x & 1023) * 4 + wave) * 16;
  const int col  = lane & 15;

  // ---- stage per-head weights into LDS (B-fragment layout) -------------
  // Expert B = W^T plus bias row: B[d][o] = W[o][d] for d<30, B[30][o]=b[o].
  // Lane l holds column o = ot*16+(l&15), k-range d = (l<16 ? 0..15 : 16..31),
  // 16 contiguous f16 per lane.
  const float* W = expert_w + (size_t)m * NEXP * DF * DF;
  const float* Bv = expert_b + (size_t)m * NEXP * DF;
  for (int i = tid; i < NEXP * 2 * 32 * 16; i += 128) {
    int e  = i >> 10;
    int r  = i & 1023;
    int ot = r >> 9;
    int r2 = r & 511;
    int l  = r2 >> 4;
    int k  = r2 & 15;
    int o  = ot * 16 + (l & 15);
    int d  = ((l < 16) ? 0 : 16) + k;
    float v = 0.0f;
    if (o < DF) {
      if (d < DF)        v = W[(e * DF + o) * DF + d];
      else if (d == DF)  v = Bv[e * DF + o];          // bias row (d==30)
    }
    lds_wB[i] = (_Float16)v;
  }
  // Router B: column e = lane&15 (e<8 valid), rows d<30 = router_w[e][d].
  const float* R = router_w + (size_t)m * NEXP * DF;
  for (int i = tid; i < 32 * 16; i += 128) {
    int l = i >> 4;
    int k = i & 15;
    int e = l & 15;
    int d = ((l < 16) ? 0 : 16) + k;
    float v = (e < NEXP && d < DF) ? R[e * DF + d] : 0.0f;
    lds_rB[i] = (_Float16)v;
  }
  __syncthreads();

  // ---- A fragment: 16 tokens x 32 features, f16; d=30 column == 1.0 ----
  // lanes 0..15 hold K=0..15 of row M=lane; lanes 16..31 hold
  // K=16..31 of row M=lane-16 (matches B fragment k order).
  const float* xrow = xs + ((size_t)m * BTOK + b0 + col) * DF;
  const int k0 = (lane < 16) ? 0 : 16;
  v16h a;
#pragma unroll
  for (int j = 0; j < 8; ++j) a[j] = (_Float16)xrow[k0 + j];        // d<=23
#pragma unroll
  for (int j = 0; j < 8; ++j) {
    int d = k0 + 8 + j;                                             // up to 31
    float v = (d < DF) ? xrow[d] : ((d == DF) ? 1.0f : 0.0f);       // bias col
    a[8 + j] = (_Float16)v;
  }
  // prefetch next block's tile of x (speculative; OOB is dropped)
  __builtin_prefetch(xrow + 4 * 16 * DF, 0, 1);

  // ---- router logits via WMMA, top-2 softmax gates ---------------------
  {
    v16h rb = ((const v16h*)lds_rB)[lane];
    v8f cz = {};
    v8f lr = __builtin_amdgcn_wmma_f32_16x16x32_f16(false, a, false, rb,
                                                    (short)0, cz, false, false);
    if (col < NEXP) {
#pragma unroll
      for (int r = 0; r < 8; ++r) {
        int t = r + ((lane >= 16) ? 8 : 0);
        lds_logits[wave][t][col] = lr[r];
      }
    }
  }
  __syncthreads();

  float g[NEXP];
#pragma unroll
  for (int e = 0; e < NEXP; ++e) g[e] = 0.0f;
  if (lane < 16) {
    float logits[NEXP];
#pragma unroll
    for (int e = 0; e < NEXP; ++e) logits[e] = lds_logits[wave][lane][e];
    int i0 = 0; float v0 = logits[0];
#pragma unroll
    for (int e = 1; e < NEXP; ++e)
      if (logits[e] > v0) { v0 = logits[e]; i0 = e; }
    int i1 = -1; float v1 = -3.402823466e38f;
#pragma unroll
    for (int e = 0; e < NEXP; ++e)
      if (e != i0 && logits[e] > v1) { v1 = logits[e]; i1 = e; }
    float e1  = __expf(v1 - v0);          // softmax over the 2 selected
    float inv = 1.0f / (1.0f + e1);
    g[i0] = inv;
    g[i1] = e1 * inv;
  }

  // ---- dense all-expert MLP on WMMA + gated exp-combine ----------------
  v8f comb0 = {};
  v8f comb1 = {};
  const v16h* bfrags = (const v16h*)lds_wB;
#pragma unroll
  for (int e = 0; e < NEXP; ++e) {
    v16h b0f = bfrags[(e * 2 + 0) * 32 + lane];
    v16h b1f = bfrags[(e * 2 + 1) * 32 + lane];
    v8f cz0 = {};
    v8f cz1 = {};
    v8f c0 = __builtin_amdgcn_wmma_f32_16x16x32_f16(false, a, false, b0f,
                                                    (short)0, cz0, false, false);
    v8f c1 = __builtin_amdgcn_wmma_f32_16x16x32_f16(false, a, false, b1f,
                                                    (short)0, cz1, false, false);
#pragma unroll
    for (int r = 0; r < 8; ++r) {
      int t = r + ((lane >= 16) ? 8 : 0);     // token row for this C slot
      float gate = __shfl(g[e], t, 32);       // gate lives in lane==token
      comb0[r] += gate * __expf(fmaxf(c0[r], 0.0f));
      comb1[r] += gate * __expf(fmaxf(c1[r], 0.0f));
    }
  }

  // ---- log(combine), transpose through LDS, coalesced store ------------
#pragma unroll
  for (int r = 0; r < 8; ++r) {
    int t = r + ((lane >= 16) ? 8 : 0);
    float a0 = comb0[r]; a0 = (a0 == 0.0f) ? EPSV : a0;
    float a1 = comb1[r]; a1 = (a1 == 0.0f) ? EPSV : a1;
    lds_stage[wave][t][col]      = __logf(a0);
    lds_stage[wave][t][16 + col] = __logf(a1);
  }
  __syncthreads();

  float* orow = out + ((size_t)m * BTOK + b0) * DF;
  for (int i = lane; i < 16 * DF; i += 32) {
    int t = i / DF, o = i - t * DF;
    orow[i] = lds_stage[wave][t][o];
  }
}

extern "C" void kernel_launch(void* const* d_in, const int* in_sizes, int n_in,
                              void* d_out, int out_size, void* d_ws, size_t ws_size,
                              hipStream_t stream) {
  (void)in_sizes; (void)n_in; (void)out_size; (void)d_ws; (void)ws_size;
  const float* xs       = (const float*)d_in[0];
  const float* router_w = (const float*)d_in[1];
  const float* expert_w = (const float*)d_in[2];
  const float* expert_b = (const float*)d_in[3];
  // d_in[4] is k (==2), baked in as top-2.
  float* out = (float*)d_out;
  dim3 grid(MHEADS * (BTOK / (4 * 16)));   // 6144 blocks
  dim3 block(128);                          // 4 wave32s
  moe_heads_kernel<<<grid, block, 0, stream>>>(xs, router_w, expert_w,
                                               expert_b, out);
}
